// Block_13297218748657
// MI455X (gfx1250) — matrix operand order repaired and checked by reading, not compiled, not run on hardware
//
#include <hip/hip_runtime.h>
#include <hip/hip_bf16.h>

// ---------------------------------------------------------------------------
// Transformer block (B=2, T=2048, D=1024, H=16, hd=64, FF=4096) on gfx1250.
// All GEMMs + attention via v_wmma_f32_16x16x32_bf16 (bf16 in, fp32 accum).
// - Weights pre-transposed to bf16 W^T[N][K]; V pre-transposed to [B,H,hd,T].
// - GEMM: 256x128 block tile, 8 waves, 4x4 WMMA tiles/wave, double-buffered.
// - LDS staging via Tensor Data Mover (tensor_load_to_lds + s_wait_tensorcnt),
//   with D# pad fields recreating the bank-conflict-avoiding LDS pitch.
//   (6-arg builtin form: amdgpu-toolchain clang-23 lane.)
// - Flash attention streams K/V (double buffered); no 536MB score tensor.
// ---------------------------------------------------------------------------

#define DMODEL 1024
#define NHEAD  16
#define HEADD  64
#define FFDIM  4096
#define SEQT   2048
#define NTOK   4096   // B*T

typedef __attribute__((ext_vector_type(16))) __bf16        bf16x16;
typedef __attribute__((ext_vector_type(8)))  float         f32x8;
typedef __attribute__((ext_vector_type(4)))  unsigned int  u32x4;
typedef __attribute__((ext_vector_type(8)))  int           i32x8;
typedef __attribute__((ext_vector_type(4)))  int           i32x4;

union FragU { bf16x16 v; u32x4 u[2]; };

#if __has_builtin(__builtin_amdgcn_tensor_load_to_lds) && \
    __has_builtin(__builtin_amdgcn_s_wait_tensorcnt)
#define USE_TDM 1
#else
#define USE_TDM 0
#endif

__device__ __forceinline__ unsigned short f2bf(float f) {
  union { float f; unsigned u; } c; c.f = f;
  unsigned r = c.u + 0x7FFFu + ((c.u >> 16) & 1u);   // round-to-nearest-even
  return (unsigned short)(r >> 16);
}

__device__ __forceinline__ f32x8 zero8() {
  f32x8 z;
#pragma unroll
  for (int i = 0; i < 8; ++i) z[i] = 0.0f;
  return z;
}

#if USE_TDM
// 2D tile DMA global->LDS via TDM. D# per ISA ch.8: group0 = {count=1,
// lds_addr, global_addr[56:0], type=2}; group1 = {data_size=2B, pad_enable,
// pad_interval, pad_amount, tensor dims (=tile dims, never OOB), tile dims,
// dim0 stride}; groups 2/3 (and the extra descriptor slot) zero for 2D tiles.
// pad codes: interval 0=2DW..7=256DW, amount 0=1DW..127=128DW.
__device__ __forceinline__ void tdm_load_tile_2d(
    void* lds_ptr, const void* gptr, unsigned tile_d0, unsigned tile_d1,
    unsigned stride_elems, unsigned pad_interval_code, unsigned pad_amount_code) {
  const unsigned lds_addr = (unsigned)(size_t)lds_ptr;   // low 32 bits = LDS offset
  const unsigned long long ga = (unsigned long long)(size_t)gptr;
  u32x4 g0;
  g0[0] = 1u;                                            // count=1, user mode
  g0[1] = lds_addr;
  g0[2] = (unsigned)ga;
  g0[3] = (unsigned)((ga >> 32) & 0x01FFFFFFu) | (2u << 30);  // addr[56:32], type=2
  i32x8 g1;
  g1[0] = (int)((1u << 16) | (1u << 20) |                // data_size=2B, pad_enable
                (pad_interval_code << 22) | (pad_amount_code << 25));
  g1[1] = (int)((tile_d0 & 0xFFFFu) << 16);              // tensor_dim0 lo16 @bit48
  g1[2] = (int)((tile_d0 >> 16) | ((tile_d1 & 0xFFFFu) << 16)); // dim0 hi / dim1 lo
  g1[3] = (int)((tile_d1 >> 16) | (tile_d0 << 16));      // dim1 hi / tile_dim0
  g1[4] = (int)(tile_d1 & 0xFFFFu);                      // tile_dim1; tile_dim2=0
  g1[5] = (int)stride_elems;                             // tensor_dim0_stride lo32
  g1[6] = 0;                                             // stride hi / dim1_stride lo
  g1[7] = 0;                                             // dim1_stride hi (unused, 2D)
  i32x4 gz4; gz4[0] = 0; gz4[1] = 0; gz4[2] = 0; gz4[3] = 0;
  i32x8 gz8;
#pragma unroll
  for (int i = 0; i < 8; ++i) gz8[i] = 0;
  __builtin_amdgcn_tensor_load_to_lds(g0, g1, gz4, gz4, gz8, 0);
}
#endif

// ---------------------------------------------------------------------------
// Transpose + convert: W[K,N] fp32 -> Wt[N,K] bf16 (32x32 tiles through LDS).
// ---------------------------------------------------------------------------
__global__ __launch_bounds__(256) void cvt_transpose_bf16(
    const float* __restrict__ in, unsigned short* __restrict__ out, int K, int N) {
  __shared__ unsigned short tile[32][34];
  const int k0 = blockIdx.y * 32;
  const int n0 = blockIdx.x * 32;
  const int r = threadIdx.x >> 5;
  const int c = threadIdx.x & 31;
#pragma unroll
  for (int i = 0; i < 4; ++i) {
    const int kk = r + i * 8;
    tile[kk][c] = f2bf(in[(size_t)(k0 + kk) * N + n0 + c]);
  }
  __syncthreads();
#pragma unroll
  for (int i = 0; i < 4; ++i) {
    const int nn = r + i * 8;
    out[(size_t)(n0 + nn) * K + k0 + c] = tile[c][nn];
  }
}

// ---------------------------------------------------------------------------
// LayerNorm over rows of length 1024, bf16 output. One 256-thread block/row.
// ---------------------------------------------------------------------------
__global__ __launch_bounds__(256) void layernorm_bf16(
    const float* __restrict__ x, const float* __restrict__ g,
    const float* __restrict__ b, unsigned short* __restrict__ out) {
  const int row = blockIdx.x;
  const float* xr = x + (size_t)row * DMODEL;
  float vals[4]; float s = 0.f, s2 = 0.f;
#pragma unroll
  for (int i = 0; i < 4; ++i) {
    vals[i] = xr[threadIdx.x + i * 256];
    s += vals[i]; s2 += vals[i] * vals[i];
  }
#pragma unroll
  for (int m = 1; m < 32; m <<= 1) { s += __shfl_xor(s, m); s2 += __shfl_xor(s2, m); }
  __shared__ float red[2][8];
  if ((threadIdx.x & 31) == 0) { red[0][threadIdx.x >> 5] = s; red[1][threadIdx.x >> 5] = s2; }
  __syncthreads();
  float t = 0.f, t2 = 0.f;
#pragma unroll
  for (int i = 0; i < 8; ++i) { t += red[0][i]; t2 += red[1][i]; }
  const float mu  = t  * (1.0f / DMODEL);
  const float var = t2 * (1.0f / DMODEL) - mu * mu;
  const float inv = rsqrtf(var + 1e-5f);
  unsigned short* orow = out + (size_t)row * DMODEL;
#pragma unroll
  for (int i = 0; i < 4; ++i) {
    const int c = threadIdx.x + i * 256;
    orow[c] = f2bf((vals[i] - mu) * inv * g[c] + b[c]);
  }
}

// ---------------------------------------------------------------------------
// bf16 GEMM: C[M,N] = act(A[M,K] @ W[K,N] + bias) (+ resid). W passed as
// Wt[N,K] bf16. 256x128 tile/WG; 8 waves (4Mx2N); 4x4 WMMA tiles per wave;
// double-buffered LDS staged by the Tensor Data Mover.
// ---------------------------------------------------------------------------
template<int GELU, int RESID, int OUTBF16>
__global__ __launch_bounds__(256) void gemm_bf16_256x128(
    const unsigned short* __restrict__ A, const unsigned short* __restrict__ Wt,
    const float* __restrict__ bias, const float* __restrict__ resid,
    void* __restrict__ Cout, int M, int N, int K) {
  __shared__ __align__(16) unsigned short As[2][256][40];
  __shared__ __align__(16) unsigned short Bs[2][128][40];

  const int tid  = threadIdx.x;
  const int lane = tid & 31;
  const int wave = tid >> 5;
  const int l16  = lane & 15;
  const int lh   = lane >> 4;
  const int wm   = wave & 3;   // M strip of 64 rows
  const int wn   = wave >> 2;  // N strip of 64 cols
  const int m0   = blockIdx.y * 256;
  const int n0   = blockIdx.x * 128;

  f32x8 acc[4][4];
#pragma unroll
  for (int mt = 0; mt < 4; ++mt)
#pragma unroll
    for (int nt = 0; nt < 4; ++nt) acc[mt][nt] = zero8();

  auto compute = [&](int p) {
    bf16x16 af[4], bfm[4];
#pragma unroll
    for (int mt = 0; mt < 4; ++mt) {  // A frag: 16-bit 16x32 lane layout
      FragU f;
      const int m   = wm * 64 + mt * 16 + l16;
      const int klo = lh * 8;
      f.u[0] = *(const u32x4*)&As[p][m][klo];
      f.u[1] = *(const u32x4*)&As[p][m][klo + 16];
      af[mt] = f.v;
    }
#pragma unroll
    for (int nt = 0; nt < 4; ++nt) {  // B frag: lane n=l16, K = 8h..8h+7, 16+8h..
      FragU f;
      const int n  = wn * 64 + nt * 16 + l16;
      const int kb = lh * 8;
      f.u[0] = *(const u32x4*)&Bs[p][n][kb];
      f.u[1] = *(const u32x4*)&Bs[p][n][kb + 16];
      bfm[nt] = f.v;
    }
#pragma unroll
    for (int mt = 0; mt < 4; ++mt)
#pragma unroll
      for (int nt = 0; nt < 4; ++nt)
        acc[mt][nt] = __builtin_amdgcn_wmma_f32_16x16x32_bf16(
            false, af[mt], false, bfm[nt], (short)0, acc[mt][nt], false, false);
  };

  const int nk = K / 32;
#if USE_TDM
  // Wave 0 drives the TDM; rows pad 64B data + 16B -> pitch 80B (=[.][40]).
  auto stage = [&](int kt, int p) {
    if (wave == 0) {
      tdm_load_tile_2d(&As[p][0][0], A  + (size_t)m0 * K + kt * 32, 32, 256,
                       (unsigned)K, 3, 3);
      tdm_load_tile_2d(&Bs[p][0][0], Wt + (size_t)n0 * K + kt * 32, 32, 128,
                       (unsigned)K, 3, 3);
    }
  };
  stage(0, 0);
  for (int kt = 0; kt < nk; ++kt) {
    const int p = kt & 1;
    if (wave == 0) __builtin_amdgcn_s_wait_tensorcnt(0);  // tile kt landed
    __syncthreads();
    if (kt + 1 < nk) stage(kt + 1, p ^ 1);                // DMA overlaps compute
    compute(p);
  }
#else
  const int brow = tid >> 1;
  const int bseg = (tid & 1) * 16;
  auto load_tiles = [&](int kt, int p) {
    const int k0 = kt * 32;
    const unsigned short* srcA = A + (size_t)(m0 + tid) * K + k0;
#pragma unroll
    for (int sg = 0; sg < 4; ++sg)
      *(u32x4*)&As[p][tid][sg * 8] = *(const u32x4*)(srcA + sg * 8);
    const unsigned short* srcB = Wt + (size_t)(n0 + brow) * K + k0 + bseg;
    *(u32x4*)&Bs[p][brow][bseg]     = *(const u32x4*)(srcB);
    *(u32x4*)&Bs[p][brow][bseg + 8] = *(const u32x4*)(srcB + 8);
  };
  load_tiles(0, 0);
  for (int kt = 0; kt < nk; ++kt) {
    const int p = kt & 1;
    __syncthreads();
    if (kt + 1 < nk) load_tiles(kt + 1, p ^ 1);
    compute(p);
  }
#endif

  // Epilogue. C layout: VGPR g, lanes0-15 -> M=g, lanes16-31 -> M=g+8, N=l16.
#pragma unroll
  for (int mt = 0; mt < 4; ++mt)
#pragma unroll
    for (int nt = 0; nt < 4; ++nt)
#pragma unroll
      for (int g = 0; g < 8; ++g) {
        const int row = m0 + wm * 64 + mt * 16 + g + 8 * lh;
        const int col = n0 + wn * 64 + nt * 16 + l16;
        float v = acc[mt][nt][g] + bias[col];
        if (GELU) v = 0.5f * v * (1.0f + erff(v * 0.70710678118f)); // exact gelu
        if (RESID) v += resid[(size_t)row * N + col];
        if (OUTBF16) ((unsigned short*)Cout)[(size_t)row * N + col] = f2bf(v);
        else         ((float*)Cout)[(size_t)row * N + col] = v;
      }
}

// ---------------------------------------------------------------------------
// qkv fp32 [NTOK,3072] -> bf16 q/k [B,H,T,64] and v TRANSPOSED [B,H,64,T].
// ---------------------------------------------------------------------------
__global__ __launch_bounds__(256) void split_qkv(
    const float* __restrict__ qkv, unsigned short* __restrict__ q,
    unsigned short* __restrict__ k, unsigned short* __restrict__ v) {
  size_t i = (size_t)blockIdx.x * 256 + threadIdx.x;   // NTOK*3072 total
  const int tok = (int)(i / (3 * DMODEL));
  const int c   = (int)(i % (3 * DMODEL));
  const int which = c / DMODEL;
  const int cc    = c % DMODEL;
  const int h = cc >> 6, d = cc & 63;
  const int b = tok >> 11, t = tok & 2047;
  const unsigned short val = f2bf(qkv[i]);
  if (which == 2) {
    v[((size_t)((b * NHEAD + h) * HEADD + d)) * SEQT + t] = val;
  } else {
    const size_t dst = (((size_t)(b * NHEAD + h) * SEQT) + t) * HEADD + d;
    if (which == 0) q[dst] = val; else k[dst] = val;
  }
}

// ---------------------------------------------------------------------------
// Flash attention: WG = 128 queries x one (b,h). 8 waves, 16 q-rows each.
// K in [T,hd]; V pre-transposed [hd,T]. K/V tiles double-buffered, staged by
// the TDM (one barrier per key tile). Online softmax with 16-lane half-wave
// shuffles. Output bf16 [B,T,H*hd] (= proj GEMM A-matrix layout).
// ---------------------------------------------------------------------------
__global__ __launch_bounds__(256) void flash_attn(
    const unsigned short* __restrict__ q, const unsigned short* __restrict__ k,
    const unsigned short* __restrict__ v, unsigned short* __restrict__ out) {
  __shared__ __align__(16) unsigned short Ks[2][64][72];   // [key][hd]
  __shared__ __align__(16) unsigned short Vs[2][64][72];   // [hd][key]
  __shared__ __align__(16) unsigned short Ps[8][16][72];   // per-wave P tile

  const int lane = threadIdx.x & 31;
  const int wave = threadIdx.x >> 5;
  const int l16  = lane & 15;
  const int lh   = lane >> 4;
  const int q0   = blockIdx.x * 128;
  const int h    = blockIdx.y;
  const int b    = blockIdx.z;
  const size_t bh = ((size_t)b * NHEAD + h) * SEQT * HEADD;
  const unsigned short* qb = q + bh;
  const unsigned short* kb = k + bh;
  const unsigned short* vb = v + bh;   // [hd][T]

  // Q fragments (A-matrix layout) for this wave's 16 rows, in registers.
  bf16x16 aq[2];
  {
    const int qrow = q0 + wave * 16 + l16;
#pragma unroll
    for (int ks = 0; ks < 2; ++ks) {
      FragU f;
      const int kbase = ks * 32 + lh * 8;
      f.u[0] = *(const u32x4*)&qb[(size_t)qrow * HEADD + kbase];
      f.u[1] = *(const u32x4*)&qb[(size_t)qrow * HEADD + kbase + 16];
      aq[ks] = f.v;
    }
  }

  f32x8 o[4];
#pragma unroll
  for (int nt = 0; nt < 4; ++nt) o[nt] = zero8();
  float m_s[8], l_s[8];
#pragma unroll
  for (int g = 0; g < 8; ++g) { m_s[g] = -1e30f; l_s[g] = 0.f; }

  const int ldrow = threadIdx.x >> 2;        // 0..63
  const int ldseg = (threadIdx.x & 3) * 16;  // 0..48
  (void)ldrow; (void)ldseg;

#if USE_TDM
  // Rows: 128B data + 16B pad -> pitch 144B (=[.][72]).
  auto stage = [&](int kt0, int p) {
    if (wave == 0) {
      tdm_load_tile_2d(&Ks[p][0][0], kb + (size_t)kt0 * HEADD, 64, 64, HEADD, 4, 3);
      tdm_load_tile_2d(&Vs[p][0][0], vb + (size_t)kt0,         64, 64, SEQT,  4, 3);
    }
  };
  stage(0, 0);
#endif

  const int niter = SEQT / 64;
  for (int it = 0; it < niter; ++it) {
    const int kt0 = it * 64;
    const int p = it & 1;
#if USE_TDM
    if (wave == 0) __builtin_amdgcn_s_wait_tensorcnt(0);
    __syncthreads();                       // tiles visible; prev reads done
    if (it + 1 < niter) stage(kt0 + 64, p ^ 1);
#else
    __syncthreads();
    {
      const unsigned short* src = kb + (size_t)(kt0 + ldrow) * HEADD + ldseg;
      *(u32x4*)&Ks[p][ldrow][ldseg]     = *(const u32x4*)(src);
      *(u32x4*)&Ks[p][ldrow][ldseg + 8] = *(const u32x4*)(src + 8);
    }
    {
      const unsigned short* src = vb + (size_t)ldrow * SEQT + kt0 + ldseg;
      *(u32x4*)&Vs[p][ldrow][ldseg]     = *(const u32x4*)(src);
      *(u32x4*)&Vs[p][ldrow][ldseg + 8] = *(const u32x4*)(src + 8);
    }
    __syncthreads();
#endif

    // S = Q @ K^T  (4 key tiles of 16, K-dim = hd, 2 steps of 32)
    f32x8 s[4];
#pragma unroll
    for (int nt = 0; nt < 4; ++nt) {
      s[nt] = zero8();
#pragma unroll
      for (int ks = 0; ks < 2; ++ks) {
        FragU fb;
        const int key = nt * 16 + l16;
        const int ko  = ks * 32 + lh * 8;
        fb.u[0] = *(const u32x4*)&Ks[p][key][ko];
        fb.u[1] = *(const u32x4*)&Ks[p][key][ko + 16];
        s[nt] = __builtin_amdgcn_wmma_f32_16x16x32_bf16(
            false, aq[ks], false, fb.v, (short)0, s[nt], false, false);
      }
    }
#pragma unroll
    for (int nt = 0; nt < 4; ++nt)
#pragma unroll
      for (int g = 0; g < 8; ++g) s[nt][g] *= 0.125f;   // 1/sqrt(64)

    // Online softmax (rows split per lane-half; reduce across 16 lanes).
    float alpha[8];
#pragma unroll
    for (int g = 0; g < 8; ++g) {
      float rmax = fmaxf(fmaxf(s[0][g], s[1][g]), fmaxf(s[2][g], s[3][g]));
      rmax = fmaxf(rmax, __shfl_xor(rmax, 1));
      rmax = fmaxf(rmax, __shfl_xor(rmax, 2));
      rmax = fmaxf(rmax, __shfl_xor(rmax, 4));
      rmax = fmaxf(rmax, __shfl_xor(rmax, 8));
      const float mnew = fmaxf(m_s[g], rmax);
      alpha[g] = __expf(m_s[g] - mnew);
      m_s[g] = mnew;
    }
    float rsum[8];
#pragma unroll
    for (int g = 0; g < 8; ++g) rsum[g] = 0.f;
#pragma unroll
    for (int nt = 0; nt < 4; ++nt)
#pragma unroll
      for (int g = 0; g < 8; ++g) {
        const float pv = __expf(s[nt][g] - m_s[g]);
        s[nt][g] = pv;
        rsum[g] += pv;
      }
#pragma unroll
    for (int g = 0; g < 8; ++g) {
      float rs = rsum[g];
      rs += __shfl_xor(rs, 1);
      rs += __shfl_xor(rs, 2);
      rs += __shfl_xor(rs, 4);
      rs += __shfl_xor(rs, 8);
      l_s[g] = l_s[g] * alpha[g] + rs;
#pragma unroll
      for (int nt = 0; nt < 4; ++nt) o[nt][g] *= alpha[g];
    }

    // P -> per-wave LDS (C layout -> A layout round trip); wave-local, in-order DS.
#pragma unroll
    for (int nt = 0; nt < 4; ++nt)
#pragma unroll
      for (int g = 0; g < 8; ++g)
        Ps[wave][g + 8 * lh][nt * 16 + l16] = f2bf(s[nt][g]);

    // O += P @ V  (K-dim = 64 keys, 2 steps of 32)
#pragma unroll
    for (int ks = 0; ks < 2; ++ks) {
      FragU fp;
      const int kbase = ks * 32 + lh * 8;
      fp.u[0] = *(const u32x4*)&Ps[wave][l16][kbase];
      fp.u[1] = *(const u32x4*)&Ps[wave][l16][kbase + 16];
#pragma unroll
      for (int nt = 0; nt < 4; ++nt) {
        FragU fv;
        const int hdn = nt * 16 + l16;
        const int ko  = ks * 32 + lh * 8;
        fv.u[0] = *(const u32x4*)&Vs[p][hdn][ko];
        fv.u[1] = *(const u32x4*)&Vs[p][hdn][ko + 16];
        o[nt] = __builtin_amdgcn_wmma_f32_16x16x32_bf16(
            false, fp.v, false, fv.v, (short)0, o[nt], false, false);
      }
    }
  }

  // Normalize and write bf16 in [B,T,H*hd] layout.
#pragma unroll
  for (int nt = 0; nt < 4; ++nt)
#pragma unroll
    for (int g = 0; g < 8; ++g) {
      const int qr = q0 + wave * 16 + g + 8 * lh;
      const float val = o[nt][g] / l_s[g];
      out[((size_t)(b * SEQT + qr)) * DMODEL + h * HEADD + nt * 16 + l16] = f2bf(val);
    }
}

// ---------------------------------------------------------------------------
extern "C" void kernel_launch(void* const* d_in, const int* in_sizes, int n_in,
                              void* d_out, int out_size, void* d_ws, size_t ws_size,
                              hipStream_t stream) {
  (void)in_sizes; (void)n_in; (void)out_size; (void)ws_size;
  const float* x      = (const float*)d_in[0];
  const float* ln1_g  = (const float*)d_in[1];
  const float* ln1_b  = (const float*)d_in[2];
  const float* ln2_g  = (const float*)d_in[3];
  const float* ln2_b  = (const float*)d_in[4];
  const float* w_qkv  = (const float*)d_in[5];
  const float* b_qkv  = (const float*)d_in[6];
  const float* w_proj = (const float*)d_in[7];
  const float* b_proj = (const float*)d_in[8];
  const float* w_fc1  = (const float*)d_in[9];
  const float* b_fc1  = (const float*)d_in[10];
  const float* w_fc2  = (const float*)d_in[11];
  const float* b_fc2  = (const float*)d_in[12];
  float* out = (float*)d_out;

  char* ws = (char*)d_ws;
  size_t off = 0;
  auto alloc = [&](size_t bytes) -> char* {
    char* p = ws + off;
    off += (bytes + 255) & ~(size_t)255;
    return p;
  };
  unsigned short* wqkvT = (unsigned short*)alloc((size_t)DMODEL * 3 * DMODEL * 2);
  unsigned short* wprjT = (unsigned short*)alloc((size_t)DMODEL * DMODEL * 2);
  unsigned short* wfc1T = (unsigned short*)alloc((size_t)DMODEL * FFDIM * 2);
  unsigned short* wfc2T = (unsigned short*)alloc((size_t)FFDIM * DMODEL * 2);
  unsigned short* h1    = (unsigned short*)alloc((size_t)NTOK * DMODEL * 2);
  float*          qkv32 = (float*)         alloc((size_t)NTOK * 3 * DMODEL * 4);
  unsigned short* qbf   = (unsigned short*)alloc((size_t)NTOK * DMODEL * 2);
  unsigned short* kbf   = (unsigned short*)alloc((size_t)NTOK * DMODEL * 2);
  unsigned short* vbf   = (unsigned short*)alloc((size_t)NTOK * DMODEL * 2);
  unsigned short* attn  = (unsigned short*)alloc((size_t)NTOK * DMODEL * 2);
  float*          x1    = (float*)         alloc((size_t)NTOK * DMODEL * 4);
  unsigned short* h2    = (unsigned short*)alloc((size_t)NTOK * DMODEL * 2);
  unsigned short* hff   = (unsigned short*)alloc((size_t)NTOK * FFDIM * 2);

  // 1) weights -> bf16, transposed to [N][K]
  cvt_transpose_bf16<<<dim3(3 * DMODEL / 32, DMODEL / 32), 256, 0, stream>>>(w_qkv, wqkvT, DMODEL, 3 * DMODEL);
  cvt_transpose_bf16<<<dim3(DMODEL / 32, DMODEL / 32), 256, 0, stream>>>(w_proj, wprjT, DMODEL, DMODEL);
  cvt_transpose_bf16<<<dim3(FFDIM / 32, DMODEL / 32), 256, 0, stream>>>(w_fc1, wfc1T, DMODEL, FFDIM);
  cvt_transpose_bf16<<<dim3(DMODEL / 32, FFDIM / 32), 256, 0, stream>>>(w_fc2, wfc2T, FFDIM, DMODEL);

  // 2) LN1 -> bf16
  layernorm_bf16<<<NTOK, 256, 0, stream>>>(x, ln1_g, ln1_b, h1);

  // 3) QKV GEMM (fp32 out) -> split to per-head bf16 (V transposed)
  gemm_bf16_256x128<0, 0, 0><<<dim3(3 * DMODEL / 128, NTOK / 256), 256, 0, stream>>>(
      h1, wqkvT, b_qkv, nullptr, qkv32, NTOK, 3 * DMODEL, DMODEL);
  split_qkv<<<(int)(((size_t)NTOK * 3 * DMODEL) / 256), 256, 0, stream>>>(qkv32, qbf, kbf, vbf);

  // 4) flash attention -> bf16 [B,T,C]
  flash_attn<<<dim3(SEQT / 128, NHEAD, 2), 256, 0, stream>>>(qbf, kbf, vbf, attn);

  // 5) proj GEMM + residual(x) -> x1 (fp32)
  gemm_bf16_256x128<0, 1, 0><<<dim3(DMODEL / 128, NTOK / 256), 256, 0, stream>>>(
      attn, wprjT, b_proj, x, x1, NTOK, DMODEL, DMODEL);

  // 6) LN2 -> bf16
  layernorm_bf16<<<NTOK, 256, 0, stream>>>(x1, ln2_g, ln2_b, h2);

  // 7) fc1 GEMM + exact GELU -> bf16
  gemm_bf16_256x128<1, 0, 1><<<dim3(FFDIM / 128, NTOK / 256), 256, 0, stream>>>(
      h2, wfc1T, b_fc1, nullptr, hff, NTOK, FFDIM, DMODEL);

  // 8) fc2 GEMM + residual(x1) -> d_out (fp32)
  gemm_bf16_256x128<0, 1, 0><<<dim3(DMODEL / 128, NTOK / 256), 256, 0, stream>>>(
      hff, wfc2T, b_fc2, x1, out, NTOK, DMODEL, FFDIM);
}
